// NoisyLinear_66314295050512
// MI455X (gfx1250) — matrix operand order repaired and checked by reading, not compiled, not run on hardware
//
#include <hip/hip_runtime.h>
#include <stdint.h>

// ---------------------------------------------------------------------------
// NoisyLinear: out[b,n,o] = sum_i x[b,n,i] * (W[i,o] + eps_w[b,i,o]*sig_w[i,o])
//                           + bias[o] + eps_b[b,o]*sig_b[o]
// B=256, N=64, IN=512, OUT=512, fp32 in/out.
// bf16 WMMA (v_wmma_f32_16x16x32_bf16), f32 accumulate, double-buffered LDS,
// software-pipelined HBM streaming of the 268 MB eps_w tensor.
// ---------------------------------------------------------------------------

#define B_      256
#define N_      64
#define IN_     512
#define OUT_    512
#define KB      32            // K chunk per iteration (one bf16 WMMA K-step)
#define TN      64            // output columns per block
#define NITER   (IN_ / KB)    // 16
#define XSTRIDE 20            // x-tile row pitch in u32 (80 B, swizzle pad)
#define WSTRIDE 20            // W-tile col pitch in u32 (80 B, swizzle pad)

typedef __attribute__((ext_vector_type(16))) __bf16 v16bf;
typedef __attribute__((ext_vector_type(8)))  float  v8f;

struct Q2 { uint4 a; uint4 b; };            // 32B fragment container
static_assert(sizeof(Q2) == sizeof(v16bf), "frag size");

// ---- f32 -> bf16, round-half-up (1 VALU/element) -------------------------
#if __has_builtin(__builtin_amdgcn_cvt_pk_bf16_f32)
typedef __attribute__((ext_vector_type(2))) __bf16 v2bf;
__device__ __forceinline__ uint32_t pack_bf16x2(float a, float b) {
    v2bf r = __builtin_amdgcn_cvt_pk_bf16_f32(a, b);   // v_cvt_pk_bf16_f32
    return __builtin_bit_cast(uint32_t, r);
}
#else
__device__ __forceinline__ uint32_t pack_bf16x2(float a, float b) {
    uint32_t ua = __builtin_bit_cast(uint32_t, a) + 0x8000u;  // v_add_nc_u32
    uint32_t ub = __builtin_bit_cast(uint32_t, b) + 0x8000u;
#if __has_builtin(__builtin_amdgcn_perm)
    return __builtin_amdgcn_perm(ub, ua, 0x07060302u);        // v_perm_b32 merge
#else
    return (ua >> 16) | (ub & 0xFFFF0000u);
#endif
}
#endif

// round-half-up f32 -> bf16 via high half (lowers to ds_store_b16_d16_hi)
__device__ __forceinline__ void store_bf16_hi(uint16_t* p, float f) {
    uint32_t u = __builtin_bit_cast(uint32_t, f) + 0x8000u;   // 1 VALU
    *p = (uint16_t)(u >> 16);
}

__device__ __forceinline__ v16bf load_frag(const uint32_t* p0, const uint32_t* p1) {
    Q2 q;
    q.a = *(const uint4*)p0;
    q.b = *(const uint4*)p1;
    return __builtin_bit_cast(v16bf, q);
}

// Per-thread register staging for the software pipeline
struct XReg { float4 a, b; };                            // 8 VGPRs
struct WReg { float4 e0, e1, w0, w1, s0, s1; };          // 24 VGPRs

__device__ __forceinline__ void fetch_x(XReg& r, const float* __restrict__ xb,
                                        int k0, int xrow, int xk) {
    const float4* p = (const float4*)(xb + (size_t)xrow * IN_ + k0 + xk);
    r.a = p[0];
    r.b = p[1];
}

// rows k0+kk and k0+kk+16, 4 consecutive columns at oc (16B aligned, coalesced)
__device__ __forceinline__ void fetch_w(WReg& r, const float* __restrict__ web,
                                        const float* __restrict__ wgl,
                                        const float* __restrict__ sw,
                                        int k0, int kk, int oc) {
    const size_t r0 = (size_t)(k0 + kk) * OUT_ + oc;
    const size_t r1 = r0 + (size_t)16 * OUT_;
    r.e0 = *(const float4*)(web + r0);
    r.e1 = *(const float4*)(web + r1);
    r.w0 = *(const float4*)(wgl + r0);
    r.w1 = *(const float4*)(wgl + r1);
    r.s0 = *(const float4*)(sw + r0);
    r.s1 = *(const float4*)(sw + r1);
}

__device__ __forceinline__ void stage(uint32_t* __restrict__ lx,
                                      uint16_t* __restrict__ lw16,
                                      const XReg& xr, const WReg& wr,
                                      int xrow, int xk, int q, int kk) {
    // x tile: row-major [row][k-pair], contiguous u32 quad -> ds_store_b128
    uint32_t* dx = &lx[xrow * XSTRIDE + (xk >> 1)];
    dx[0] = pack_bf16x2(xr.a.x, xr.a.y);
    dx[1] = pack_bf16x2(xr.a.z, xr.a.w);
    dx[2] = pack_bf16x2(xr.b.x, xr.b.y);
    dx[3] = pack_bf16x2(xr.b.z, xr.b.w);

    // noisy W tile: transposed [col][k], 16-bit high-half stores
    const float nv0[4] = { __builtin_fmaf(wr.e0.x, wr.s0.x, wr.w0.x),
                           __builtin_fmaf(wr.e0.y, wr.s0.y, wr.w0.y),
                           __builtin_fmaf(wr.e0.z, wr.s0.z, wr.w0.z),
                           __builtin_fmaf(wr.e0.w, wr.s0.w, wr.w0.w) };
    const float nv1[4] = { __builtin_fmaf(wr.e1.x, wr.s1.x, wr.w1.x),
                           __builtin_fmaf(wr.e1.y, wr.s1.y, wr.w1.y),
                           __builtin_fmaf(wr.e1.z, wr.s1.z, wr.w1.z),
                           __builtin_fmaf(wr.e1.w, wr.s1.w, wr.w1.w) };
#pragma unroll
    for (int i = 0; i < 4; ++i) {
        const int col = 4 * q + i;
        store_bf16_hi(&lw16[col * (2 * WSTRIDE) + kk],      nv0[i]);   // K = kk
        store_bf16_hi(&lw16[col * (2 * WSTRIDE) + kk + 16], nv1[i]);   // K = kk+16
    }
}

__global__ __launch_bounds__(256)
void noisy_linear_wmma(const float* __restrict__ x,
                       const float* __restrict__ w,
                       const float* __restrict__ bias,
                       const float* __restrict__ sw,
                       const float* __restrict__ sb,
                       const float* __restrict__ we,
                       const float* __restrict__ be,
                       float* __restrict__ out)
{
    __shared__ __align__(16) uint32_t lds_x[2][N_ * XSTRIDE];      // 2 x 5 KB
    __shared__ __align__(16) uint16_t lds_w[2][TN * WSTRIDE * 2];  // 2 x 5 KB

    const int b    = blockIdx.y;
    const int oc0  = blockIdx.x * TN;
    const int t    = threadIdx.x;
    const int lane = t & 31;
    const int wv   = t >> 5;                    // wave id 0..7

    // wave -> (row tile, col pair) mapping of the 64x64 output block
    const int rt = (wv & 3) * 16;               // rows rt..rt+15
    const int ct = (wv >> 2) * 32;              // cols ct..ct+31 (two 16-wide tiles)

    v8f acc0 = {};
    v8f acc1 = {};

    const float* xb  = x  + (size_t)b * N_  * IN_;
    const float* web = we + (size_t)b * IN_ * OUT_;

    // staging maps
    const int xrow = t >> 2;                    // 0..63
    const int xk   = (t & 3) * 8;               // 0,8,16,24
    const int q    = t & 15;                    // column quad -> cols 4q..4q+3
    const int kk   = t >> 4;                    // 0..15 -> rows kk, kk+16
    const int ocw  = oc0 + 4 * q;

    // fragment lane coordinates (ISA 7.12.2 bf16 layouts, wave32)
    const int fm = lane & 15;                   // A row / B col within tile
    const int kh = lane >> 4;                   // lane-half selects K group

    // ---- software pipeline: preload chunk 0 ----
    XReg xr;
    WReg wr;
    fetch_x(xr, xb, 0, xrow, xk);
    fetch_w(wr, web, w, sw, 0, kk, ocw);

#pragma unroll 2
    for (int it = 0; it < NITER; ++it) {
        const int buf = it & 1;
        // commit current chunk to LDS (regs -> bf16)
        stage(lds_x[buf], lds_w[buf], xr, wr, xrow, xk, q, kk);

        // issue next chunk's global loads; they stream from HBM while this
        // chunk's WMMAs run (only consumed after the next barrier)
        const int kn = (it + 1) * KB;
        if (it + 1 < NITER) {
            fetch_x(xr, xb, kn, xrow, xk);
            fetch_w(wr, web, w, sw, kn, kk, ocw);
            if (it + 2 < NITER)   // pull chunk i+2 toward L2 (global_prefetch_b8)
                __builtin_prefetch(web + (size_t)(kn + KB + kk) * OUT_ + ocw, 0, 1);
        }

        __syncthreads();

        // ---- WMMA: D = A(16x32 bf16) x B(32x16 bf16) + C(f32) ----
        // A: lanes 0-15 hold K 0-7 & 16-23; lanes 16-31 hold K 8-15 & 24-31
        const uint32_t* ax = &lds_x[buf][(rt + fm) * XSTRIDE + kh * 4];
        v16bf A = load_frag(ax, ax + 8);
        // B: lanes 0-15 hold K 0-7 & 16-23; lanes 16-31 hold K 8-15 & 24-31, column = fm
        const uint32_t* b0 =
            (const uint32_t*)&lds_w[buf][(ct + fm) * (2 * WSTRIDE)] + kh * 4;
        v16bf B0 = load_frag(b0, b0 + 8);
        const uint32_t* b1 =
            (const uint32_t*)&lds_w[buf][(ct + 16 + fm) * (2 * WSTRIDE)] + kh * 4;
        v16bf B1 = load_frag(b1, b1 + 8);

        acc0 = __builtin_amdgcn_wmma_f32_16x16x32_bf16(
                   false, A, false, B0, (short)0, acc0, false, false);
        acc1 = __builtin_amdgcn_wmma_f32_16x16x32_bf16(
                   false, A, false, B1, (short)0, acc1, false, false);
    }

    // ---- epilogue: + bias + eps_b * sigma_b, store f32 ----
    {
        const int c0 = oc0 + ct + fm;           // column for acc0
        const int c1 = c0 + 16;                 // column for acc1
        const float bc0 = __builtin_fmaf(be[(size_t)b * OUT_ + c0], sb[c0], bias[c0]);
        const float bc1 = __builtin_fmaf(be[(size_t)b * OUT_ + c1], sb[c1], bias[c1]);

        // C/D layout: VGPR j -> row rt + j + 8*(lane>=16), col = fm (per tile)
        float* ob = out + ((size_t)b * N_ + rt + kh * 8) * OUT_;
#pragma unroll
        for (int j = 0; j < 8; ++j) {
            ob[(size_t)j * OUT_ + c0] = acc0[j] + bc0;
            ob[(size_t)j * OUT_ + c1] = acc1[j] + bc1;
        }
    }
}

extern "C" void kernel_launch(void* const* d_in, const int* in_sizes, int n_in,
                              void* d_out, int out_size, void* d_ws, size_t ws_size,
                              hipStream_t stream) {
    const float* x    = (const float*)d_in[0];   // (B, N, IN)
    const float* w    = (const float*)d_in[1];   // (IN, OUT)
    const float* bias = (const float*)d_in[2];   // (1, OUT)
    const float* sw   = (const float*)d_in[3];   // (IN, OUT)
    const float* sb   = (const float*)d_in[4];   // (1, OUT)
    const float* we   = (const float*)d_in[5];   // (B, IN, OUT)
    const float* be   = (const float*)d_in[6];   // (B, 1, OUT)

    dim3 grid(OUT_ / TN, B_);                    // (8, 256)
    noisy_linear_wmma<<<grid, 256, 0, stream>>>(x, w, bias, sw, sb, we, be,
                                                (float*)d_out);
}
